// JointSelfAttention_68341519614818
// MI455X (gfx1250) — compile-verified, matrix operand order checked by reading, not run on hardware
//
#include <hip/hip_runtime.h>
#include <cmath>
#include <cstdint>

// ---------------------------------------------------------------------------
// MI455X / gfx1250 (CDNA5, wave32) fused causal self-attention block.
// Matrix math on v_wmma_f32_16x16x32_bf16; tile staging via CDNA5
// global_load_async_to_lds_b128 (ASYNCcnt) where the copy is layout-preserving.
// ---------------------------------------------------------------------------

typedef __attribute__((ext_vector_type(16))) __bf16 v16bf;
typedef __attribute__((ext_vector_type(8)))  float  v8f;

union FragBF { v16bf f; uint4 q[2]; };

static constexpr int Bn  = 4;
static constexpr int Tn  = 2048;
static constexpr int Cn  = 1024;
static constexpr int Hn  = 16;
static constexpr int HDn = 64;

// CDNA5 async global->LDS copy (16B per lane), tracked by ASYNCcnt.
__device__ __forceinline__ void async_b128(unsigned lds_off, const void* gptr) {
#if defined(__gfx1250__)
  asm volatile("global_load_async_to_lds_b128 %0, %1, off"
               :: "v"(lds_off), "v"(gptr) : "memory");
#endif
}
__device__ __forceinline__ void wait_async0() {
#if defined(__gfx1250__)
  asm volatile("s_wait_asynccnt 0" ::: "memory");
#endif
}
__device__ __forceinline__ void wait_ds0() {
#if defined(__gfx1250__)
  asm volatile("s_wait_dscnt 0" ::: "memory");
#endif
}

// ------------------------------- convert -----------------------------------
__global__ void f32_to_bf16_kernel(const float* __restrict__ in,
                                   __bf16* __restrict__ out, int n) {
  int i = blockIdx.x * blockDim.x + threadIdx.x;
  if (i < n) out[i] = (__bf16)in[i];
}

// ------------------------------- GEMM --------------------------------------
// D[M,N] = A[M,K] @ W[N,K]^T   (A, W bf16 row-major; D fp32 or bf16)
// 128 threads = 4 waves; block tile 128x64; wave tile 32(M) x 64(N); K step 32.
// Per wave per K-step: 8 WMMAs from 2 A-fragments x 4 W-fragments.
template <bool OUT_BF16>
__global__ __launch_bounds__(128) void gemm_wmma_bf16(
    const __bf16* __restrict__ A, const __bf16* __restrict__ W,
    void* __restrict__ Dp, int M, int N, int K) {
  const int tid    = threadIdx.x;
  const int wave   = tid >> 5;
  const int lane   = tid & 31;
  const int laneHi = lane >> 4;
  const int lane16 = lane & 15;
  const int col0   = blockIdx.x * 64;
  const int row0   = blockIdx.y * 128;

  __shared__ __align__(16) __bf16 Ash[128 * 32];
  __shared__ __align__(16) __bf16 Wsh[64 * 32];
  const unsigned ashBase = (unsigned)(uintptr_t)&Ash[0];
  const unsigned wshBase = (unsigned)(uintptr_t)&Wsh[0];

  v8f acc[2][4];
#pragma unroll
  for (int mt = 0; mt < 2; ++mt)
#pragma unroll
    for (int t = 0; t < 4; ++t)
#pragma unroll
      for (int i = 0; i < 8; ++i) acc[mt][t][i] = 0.0f;

  for (int k0 = 0; k0 < K; k0 += 32) {
    __syncthreads();
    // A tile: 128 rows x 32 cols = 512 x b128 -> 4 per thread (async to LDS).
#pragma unroll
    for (int i = 0; i < 4; ++i) {
      int u = tid * 4 + i;
      int r = u >> 2, qd = u & 3;
      async_b128(ashBase + (unsigned)(r * 32 + qd * 8) * 2,
                 A + (size_t)(row0 + r) * K + k0 + qd * 8);
    }
    // W tile: 64 rows x 32 cols = 256 x b128 -> 2 per thread (async to LDS).
#pragma unroll
    for (int i = 0; i < 2; ++i) {
      int u = tid * 2 + i;
      int r = u >> 2, qd = u & 3;
      async_b128(wshBase + (unsigned)(r * 32 + qd * 8) * 2,
                 W + (size_t)(col0 + r) * K + k0 + qd * 8);
    }
    wait_async0();
    __syncthreads();

    FragBF af[2];
#pragma unroll
    for (int mt = 0; mt < 2; ++mt) {
      const __bf16* p = &Ash[(wave * 32 + mt * 16 + lane16) * 32 + laneHi * 8];
      af[mt].q[0] = *(const uint4*)p;
      af[mt].q[1] = *(const uint4*)(p + 16);
    }
#pragma unroll
    for (int t = 0; t < 4; ++t) {
      FragBF wf;
      const __bf16* p = &Wsh[(t * 16 + lane16) * 32 + laneHi * 8];
      wf.q[0] = *(const uint4*)p;
      wf.q[1] = *(const uint4*)(p + 16);
#pragma unroll
      for (int mt = 0; mt < 2; ++mt)
        acc[mt][t] = __builtin_amdgcn_wmma_f32_16x16x32_bf16(
            false, af[mt].f, false, wf.f, (short)0, acc[mt][t], false, false);
    }
  }

#pragma unroll
  for (int mt = 0; mt < 2; ++mt)
#pragma unroll
    for (int t = 0; t < 4; ++t)
#pragma unroll
      for (int r = 0; r < 8; ++r) {
        int m = row0 + wave * 32 + mt * 16 + r + 8 * laneHi;
        int n = col0 + t * 16 + lane16;
        if constexpr (OUT_BF16)
          ((__bf16*)Dp)[(size_t)m * N + n] = (__bf16)acc[mt][t][r];
        else
          ((float*)Dp)[(size_t)m * N + n] = acc[mt][t][r];
      }
}

// --------------------------- RMS-norm + RoPE -------------------------------
__global__ __launch_bounds__(256) void normrope_kernel(__bf16* __restrict__ qkv) {
  const int grp  = threadIdx.x >> 6;   // 0..3
  const int d    = threadIdx.x & 63;
  const int wave = threadIdx.x >> 5;   // 0..7
  const int gid  = blockIdx.x * 4 + grp;  // (b*T + t)*H + h
  const int h  = gid & (Hn - 1);
  const int bt = gid >> 4;
  const int t  = bt & (Tn - 1);
  __bf16* base = qkv + (size_t)bt * (3 * Cn) + h * HDn;

  __shared__ float partial[8];
  __shared__ float xbuf[4][64];

  const int i = d & 31;
  const float ang = (float)t * expf(-(float)(2 * i) * (9.210340371976184f / 64.0f));
  const float cv = (float)(__bf16)cosf(ang);  // mimic reference bf16 tables
  const float sv = (float)(__bf16)sinf(ang);

#pragma unroll
  for (int part = 0; part < 2; ++part) {  // 0 = q, 1 = k
    __bf16* ptr = base + part * Cn;
    float xv = (float)ptr[d];
    float sq = xv * xv;
#pragma unroll
    for (int msk = 16; msk >= 1; msk >>= 1) sq += __shfl_xor(sq, msk);
    if ((threadIdx.x & 31) == 0) partial[wave] = sq;
    __syncthreads();
    float ssum = partial[grp * 2] + partial[grp * 2 + 1];
    float rn = rsqrtf(ssum * (1.0f / 64.0f) + 1.1920929e-7f);
    xbuf[grp][d] = xv * rn;
    __syncthreads();
    float x1 = xbuf[grp][i];
    float x2 = xbuf[grp][i + 32];
    float y = (d < 32) ? (x1 * cv + x2 * sv) : (x2 * cv - x1 * sv);
    ptr[d] = (__bf16)y;
    __syncthreads();
  }
}

// --------------------------- flash attention -------------------------------
// Block = 4 waves, 64 query rows per (b, h); 32-key steps; causal; online
// softmax. Per step per wave: 4 WMMAs (S) + 4 WMMAs (P@V).
__global__ __launch_bounds__(128) void attn_kernel(
    const __bf16* __restrict__ qkv, __bf16* __restrict__ y) {
  const int qblk = blockIdx.x, h = blockIdx.y, b = blockIdx.z;
  const int tid    = threadIdx.x;
  const int wave   = tid >> 5;
  const int lane   = tid & 31;
  const int laneHi = lane >> 4;
  const int lane16 = lane & 15;
  const int q0     = qblk * 64;
  const int qrow0  = q0 + wave * 16;

  __shared__ __align__(16) __bf16 Ksh[32 * HDn];     // [key][hd]
  __shared__ __align__(16) __bf16 VTsh[HDn * 32];    // [hd][key] (transposed)
  __shared__ __align__(16) __bf16 Psh[4][16 * 32];   // per-wave P tile
  const unsigned kshBase = (unsigned)(uintptr_t)&Ksh[0];

  const size_t ld = 3 * Cn;
  const __bf16* kbase = qkv + (size_t)b * Tn * ld + Cn + h * HDn;
  const __bf16* vbase = qkv + (size_t)b * Tn * ld + 2 * Cn + h * HDn;

  // Q fragments: 16 rows x 64 hd, two K=32 chunks (A-matrix layout).
  FragBF qf[2];
  {
    const __bf16* p = qkv + (size_t)b * Tn * ld + (size_t)(qrow0 + lane16) * ld +
                      h * HDn + laneHi * 8;
#pragma unroll
    for (int c = 0; c < 2; ++c) {
      qf[c].q[0] = *(const uint4*)(p + c * 32);
      qf[c].q[1] = *(const uint4*)(p + c * 32 + 16);
    }
  }

  v8f o[4];
  float rowmax[8], rowsum[8];
#pragma unroll
  for (int t = 0; t < 4; ++t)
#pragma unroll
    for (int i = 0; i < 8; ++i) o[t][i] = 0.0f;
#pragma unroll
  for (int r = 0; r < 8; ++r) { rowmax[r] = -3.0e38f; rowsum[r] = 0.0f; }

  const float scale = 0.125f;  // 1/sqrt(64)
  const int nkb = min(q0 / 32 + 2, Tn / 32);

  for (int kb = 0; kb < nkb; ++kb) {
    __syncthreads();
    // K tile: layout-preserving -> async global->LDS (ASYNCcnt path).
    // V tile: needs transpose -> through VGPRs with scalar b16 scatter.
#pragma unroll
    for (int i = 0; i < 2; ++i) {
      int u = tid * 2 + i;  // 0..255 : 32 rows x 8 uint4
      int row = u >> 3, qd = u & 7;
      int kpos = kb * 32 + row;
      async_b128(kshBase + (unsigned)(row * HDn + qd * 8) * 2,
                 kbase + (size_t)kpos * ld + qd * 8);
      uint4 vd = *(const uint4*)(vbase + (size_t)kpos * ld + qd * 8);
      const __bf16* ve = (const __bf16*)&vd;
#pragma unroll
      for (int j = 0; j < 8; ++j) VTsh[(qd * 8 + j) * 32 + row] = ve[j];
    }
    wait_async0();
    __syncthreads();

    // S = Q @ K^T : two 16x16 key tiles, reduction over hd=64 (2 x K=32).
    v8f s[2];
#pragma unroll
    for (int t = 0; t < 2; ++t) {
      v8f sa;
#pragma unroll
      for (int i = 0; i < 8; ++i) sa[i] = 0.0f;
#pragma unroll
      for (int c = 0; c < 2; ++c) {
        FragBF kf;
        const __bf16* p = &Ksh[(t * 16 + lane16) * HDn + c * 32 + laneHi * 8];
        kf.q[0] = *(const uint4*)p;
        kf.q[1] = *(const uint4*)(p + 16);
        sa = __builtin_amdgcn_wmma_f32_16x16x32_bf16(
            false, qf[c].f, false, kf.f, (short)0, sa, false, false);
      }
      s[t] = sa;
    }

    // Causal mask + scale + online softmax (rows live at vgpr r, half laneHi).
#pragma unroll
    for (int r = 0; r < 8; ++r) {
      int qpos = qrow0 + r + 8 * laneHi;
      int kp   = kb * 32 + lane16;
      float v0 = (kp      <= qpos) ? s[0][r] * scale : -3.0e38f;
      float v1 = (kp + 16 <= qpos) ? s[1][r] * scale : -3.0e38f;
      float lm = fmaxf(v0, v1);
#pragma unroll
      for (int msk = 8; msk >= 1; msk >>= 1) lm = fmaxf(lm, __shfl_xor(lm, msk));
      float newm = fmaxf(rowmax[r], lm);
      float corr = expf(rowmax[r] - newm);
      rowmax[r] = newm;
      float e0 = expf(v0 - newm);
      float e1 = expf(v1 - newm);
      float ls = e0 + e1;
#pragma unroll
      for (int msk = 8; msk >= 1; msk >>= 1) ls += __shfl_xor(ls, msk);
      rowsum[r] = rowsum[r] * corr + ls;
#pragma unroll
      for (int t = 0; t < 4; ++t) o[t][r] *= corr;
      int mrow = r + 8 * laneHi;
      Psh[wave][mrow * 32 + lane16]      = (__bf16)e0;
      Psh[wave][mrow * 32 + 16 + lane16] = (__bf16)e1;
    }

    // Same-wave LDS write -> read: order with CDNA5 split DS counter.
    wait_ds0();

    // O += P(16x32) @ V(32x64)
    FragBF pf;
    {
      const __bf16* p = &Psh[wave][lane16 * 32 + laneHi * 8];
      pf.q[0] = *(const uint4*)p;
      pf.q[1] = *(const uint4*)(p + 16);
    }
#pragma unroll
    for (int t = 0; t < 4; ++t) {
      FragBF vf;
      const __bf16* p = &VTsh[(t * 16 + lane16) * 32 + laneHi * 8];
      vf.q[0] = *(const uint4*)p;
      vf.q[1] = *(const uint4*)(p + 16);
      o[t] = __builtin_amdgcn_wmma_f32_16x16x32_bf16(
          false, pf.f, false, vf.f, (short)0, o[t], false, false);
    }
  }

  // Epilogue: normalize and write y[b, qpos, h*64 + n] as bf16.
#pragma unroll
  for (int t = 0; t < 4; ++t)
#pragma unroll
    for (int r = 0; r < 8; ++r) {
      int qpos = qrow0 + r + 8 * laneHi;
      float val = o[t][r] / rowsum[r];
      y[((size_t)b * Tn + qpos) * Cn + h * HDn + t * 16 + lane16] = (__bf16)val;
    }
}

// ------------------------------- launch ------------------------------------
extern "C" void kernel_launch(void* const* d_in, const int* in_sizes, int n_in,
                              void* d_out, int out_size, void* d_ws, size_t ws_size,
                              hipStream_t stream) {
  (void)in_sizes; (void)n_in; (void)out_size; (void)ws_size;
  const float* x  = (const float*)d_in[0];   // [B,T,C]
  const float* wa = (const float*)d_in[1];   // [3C,C]
  const float* wp = (const float*)d_in[2];   // [C,C]
  float* out = (float*)d_out;                // [B,T,C] fp32

  const int M = Bn * Tn;  // 8192
  char* ws = (char*)d_ws;
  __bf16* xb   = (__bf16*)ws; ws += (size_t)M * Cn * 2;
  __bf16* wab  = (__bf16*)ws; ws += (size_t)3 * Cn * Cn * 2;
  __bf16* wpb  = (__bf16*)ws; ws += (size_t)Cn * Cn * 2;
  __bf16* qkvb = (__bf16*)ws; ws += (size_t)M * 3 * Cn * 2;
  __bf16* yb   = (__bf16*)ws; ws += (size_t)M * Cn * 2;

  {
    int n = M * Cn;
    f32_to_bf16_kernel<<<(n + 255) / 256, 256, 0, stream>>>(x, xb, n);
    n = 3 * Cn * Cn;
    f32_to_bf16_kernel<<<(n + 255) / 256, 256, 0, stream>>>(wa, wab, n);
    n = Cn * Cn;
    f32_to_bf16_kernel<<<(n + 255) / 256, 256, 0, stream>>>(wp, wpb, n);
  }

  // qkv = x @ w_attn^T  (bf16 out)
  gemm_wmma_bf16<true><<<dim3(3 * Cn / 64, M / 128), 128, 0, stream>>>(
      xb, wab, qkvb, M, 3 * Cn, Cn);

  // in-place RMS-norm + rotary on q,k
  normrope_kernel<<<M * Hn / 4, 256, 0, stream>>>(qkvb);

  // flash attention -> y (bf16)
  attn_kernel<<<dim3(Tn / 64, Hn, Bn), 128, 0, stream>>>(qkvb, yb);

  // out = y @ w_proj^T  (fp32 out)
  gemm_wmma_bf16<false><<<dim3(Cn / 64, M / 128), 128, 0, stream>>>(
      yb, wpb, out, M, Cn, Cn);
}
